// Model_58935541236393
// MI455X (gfx1250) — matrix operand order repaired and checked by reading, not compiled, not run on hardware
//
#include <hip/hip_runtime.h>
#include <hip/hip_bf16.h>

// ---------------------------------------------------------------------------
// Model dims: B=16, FRAMES=243, S=17, D=256, H=8, NTK=81, ADF=81, T=162,
// HALF=128, CH=16, DFF=1024.  Tokens = B*T*S = 44064.  x1/x2 tokens = 22032.
// ---------------------------------------------------------------------------

typedef __attribute__((ext_vector_type(16))) _Float16 v16h;
typedef __attribute__((ext_vector_type(8)))  float    v8f;

__device__ __constant__ int d_PART[17] = {0,1,1,1,2,2,2,0,0,0,0,3,3,3,4,4,4};

__device__ __forceinline__ float gelu_f(float x) {
    return 0.5f * x * (1.0f + erff(x * 0.70710678118654752440f));
}

// ---------------------------------------------------------------------------
// Constant-matrix generation.  DCT_K / IDCT_K are built zero-padded to 96x96
// and head_pre_w to 256x192 so the GEMM never needs load guards: padding
// columns are 0, killing the (finite) garbage from out-of-range B rows.
// ---------------------------------------------------------------------------
__global__ void k_const_init(const float* __restrict__ hp,
                             float* dctf, float* dctk, float* idctk,
                             float* hpre) {
    int i = blockIdx.x * blockDim.x + threadIdx.x;
    const double PI = 3.14159265358979323846;
    if (i < 81 * 243) {                       // DCT_F_TRUNC[k][n] (unpadded)
        int k = i / 243, n = i % 243;
        dctf[i] = (float)(2.0 * cos(PI * (2 * n + 1) * k / (2.0 * 243.0)));
        return;
    }
    i -= 81 * 243;
    if (i < 96 * 96) {                        // DCT_K[q][t], padded
        int q = i / 96, t = i % 96;
        dctk[i] = (q < 81 && t < 81)
            ? (float)(2.0 * cos(PI * (2 * t + 1) * q / (2.0 * 81.0))) : 0.f;
        return;
    }
    i -= 96 * 96;
    if (i < 96 * 96) {                        // IDCT_K[n][k], padded, col0 halved
        int n = i / 96, k = i % 96;
        float v = 0.f;
        if (n < 81 && k < 81) {
            double d = cos(PI * k * (2 * n + 1) / (2.0 * 81.0)) / 81.0;
            if (k == 0) d *= 0.5;
            v = (float)d;
        }
        idctk[i] = v;
        return;
    }
    i -= 96 * 96;
    if (i < 256 * 192) {                      // head_pre_w (243x162), padded
        int m = i / 192, k = i % 192;
        hpre[i] = (m < 243 && k < 162) ? hp[m * 162 + k] : 0.f;
    }
}

// ---------------------------------------------------------------------------
// xc = concat(x[:, ::3], DCT_F_TRUNC @ frames)     -> (B, 162, 17, 2)
// ---------------------------------------------------------------------------
__global__ void k_build_xc(const float* __restrict__ x,
                           const float* __restrict__ dctf,
                           float* __restrict__ xc) {
    int i = blockIdx.x * blockDim.x + threadIdx.x;
    if (i >= 44064 * 2) return;
    int c = i & 1;
    long token = i >> 1;
    int s = (int)(token % 17);
    long bt = token / 17;
    int t = (int)(bt % 162);
    int b = (int)(bt / 162);
    float val;
    if (t < 81) {
        val = x[(((long)b * 243 + 3 * t) * 17 + s) * 2 + c];
    } else {
        int k = t - 81;
        float acc = 0.f;
        for (int f = 0; f < 243; ++f)
            acc += x[(((long)b * 243 + f) * 17 + s) * 2 + c] * dctf[k * 243 + f];
        val = acc;
    }
    xc[i] = val;
}

// ---------------------------------------------------------------------------
// y = gelu(xc @ pose_emb_w.T)   (K=2, plain VALU)
// ---------------------------------------------------------------------------
__global__ void k_embed(const float* __restrict__ xc,
                        const float* __restrict__ pw,
                        float* __restrict__ y) {
    long i = (long)blockIdx.x * blockDim.x + threadIdx.x;
    if (i >= (long)44064 * 256) return;
    long token = i >> 8;
    int d = (int)(i & 255);
    float v = xc[token * 2] * pw[d * 2] + xc[token * 2 + 1] * pw[d * 2 + 1];
    y[i] = gelu_f(v);
}

// ---------------------------------------------------------------------------
// LayerNorm over last dim (256).
//   mode 0: all 44064 tokens, write at same token index.
//   mode 1: x1 rows (t<81)   -> write COMPACT rows [0,22032).
//   mode 2: x2 rows (t>=81)  -> write COMPACT rows [0,22032).
// ---------------------------------------------------------------------------
__global__ void k_ln(const float* __restrict__ X, float* __restrict__ Y,
                     const float* __restrict__ g, const float* __restrict__ bta,
                     int mode) {
    __shared__ float red[256];
    int blk = blockIdx.x;
    long token;
    if (mode == 0) token = blk;
    else {
        int b = blk / 1377, t = blk % 1377;
        token = (long)b * 2754 + (mode == 2 ? 1377 : 0) + t;
    }
    int d = threadIdx.x;
    float x = X[token * 256 + d];
    red[d] = x; __syncthreads();
    for (int s = 128; s > 0; s >>= 1) { if (d < s) red[d] += red[d + s]; __syncthreads(); }
    float m = red[0] * (1.f / 256.f);
    __syncthreads();
    float c = x - m;
    red[d] = c * c; __syncthreads();
    for (int s = 128; s > 0; s >>= 1) { if (d < s) red[d] += red[d + s]; __syncthreads(); }
    float var = red[0] * (1.f / 256.f);
    long otoken = (mode == 0) ? token : (long)blk;
    Y[otoken * 256 + d] = c * rsqrtf(var + 1e-5f) * g[d] + bta[d];
}

// ---------------------------------------------------------------------------
// Fragment loaders (all unconditional).
// A 16x32 f16 layout: lane<16 row M=lane, K in {kbase+0..15} (contiguous).
// ---------------------------------------------------------------------------
__device__ __forceinline__ v16h load_a_frag(const float* __restrict__ ar) {
    float4 a0 = *(const float4*)(ar);
    float4 a1 = *(const float4*)(ar + 4);
    float4 a2 = *(const float4*)(ar + 8);
    float4 a3 = *(const float4*)(ar + 12);
    v16h af;
    af[0] = (_Float16)a0.x; af[1] = (_Float16)a0.y;
    af[2] = (_Float16)a0.z; af[3] = (_Float16)a0.w;
    af[4] = (_Float16)a1.x; af[5] = (_Float16)a1.y;
    af[6] = (_Float16)a1.z; af[7] = (_Float16)a1.w;
    af[8]  = (_Float16)a2.x; af[9]  = (_Float16)a2.y;
    af[10] = (_Float16)a2.z; af[11] = (_Float16)a2.w;
    af[12] = (_Float16)a3.x; af[13] = (_Float16)a3.y;
    af[14] = (_Float16)a3.z; af[15] = (_Float16)a3.w;
    return af;
}

__device__ __forceinline__ v16h load_b_frag_t(const float* __restrict__ br) {
    // 16 contiguous K values from a weight row (transB): K = koff+0..15
    float4 b0 = *(const float4*)(br);
    float4 b1 = *(const float4*)(br + 4);
    float4 b2 = *(const float4*)(br + 8);
    float4 b3 = *(const float4*)(br + 12);
    v16h bf;
    bf[0] = (_Float16)b0.x; bf[1] = (_Float16)b0.y;
    bf[2] = (_Float16)b0.z; bf[3] = (_Float16)b0.w;
    bf[4] = (_Float16)b1.x; bf[5] = (_Float16)b1.y;
    bf[6] = (_Float16)b1.z; bf[7] = (_Float16)b1.w;
    bf[8]  = (_Float16)b2.x; bf[9]  = (_Float16)b2.y;
    bf[10] = (_Float16)b2.z; bf[11] = (_Float16)b2.w;
    bf[12] = (_Float16)b3.x; bf[13] = (_Float16)b3.y;
    bf[14] = (_Float16)b3.z; bf[15] = (_Float16)b3.w;
    return bf;
}

__device__ __forceinline__ v16h load_b_frag_n(const float* __restrict__ Bb,
                                              long ldB, int kk0, int col) {
    v16h bf;
#pragma unroll
    for (int e = 0; e < 16; ++e)
        bf[e] = (_Float16)Bb[(long)(kk0 + e) * ldB + col];
    return bf;
}

// ---------------------------------------------------------------------------
// Batched WMMA GEMM:  C[b] = act(A[b] * B[b] + bias) + R[b]
//   A: Mpad x Kpad row-major (padded so all loads are in-bounds; padding
//      columns k>=K are ZERO for unaligned-K operands).
//   TRANSB=1: B stored (N x K) row-major (weights, K%32==0 always).
//   TRANSB=0: B row-major K x N inside the workspace (over-reads safe).
//   One wave per 16(M) x 64(N) strip, 4 accumulators, store-guard m<M only.
//   Kpad % 32 == 0, N % 64 == 0.
// ---------------------------------------------------------------------------
template <int TRANSB>
__global__ void k_gemm(const float* __restrict__ A, long ldA, long sA,
                       const float* __restrict__ Bm, long ldB, long sB,
                       const float* __restrict__ bias,
                       const float* __restrict__ R, long ldR, long sR,
                       float* __restrict__ C, long ldC, long sC,
                       int M, int N, int Kpad, int act) {
    int lane = threadIdx.x;
    int m0 = blockIdx.x * 16;
    int n0 = blockIdx.y * 64;
    long b = blockIdx.z;
    const float* Ab = A + b * sA;
    const float* Bb = Bm + b * sB;

    int row   = m0 + (lane & 15);
    int kbase = (lane >> 4) * 16;   // A-frag K base
    int colL  = (lane & 15);
    int koff  = (lane >> 4) * 16;   // B-frag K base

    v8f acc0 = {}, acc1 = {}, acc2 = {}, acc3 = {};

    for (int k0 = 0; k0 < Kpad; k0 += 32) {
        v16h af = load_a_frag(Ab + (long)row * ldA + k0 + kbase);
        v16h bf0, bf1, bf2, bf3;
        if (TRANSB) {
            const float* br = Bb + k0 + koff;
            bf0 = load_b_frag_t(br + (long)(n0 +  0 + colL) * ldB);
            bf1 = load_b_frag_t(br + (long)(n0 + 16 + colL) * ldB);
            bf2 = load_b_frag_t(br + (long)(n0 + 32 + colL) * ldB);
            bf3 = load_b_frag_t(br + (long)(n0 + 48 + colL) * ldB);
        } else {
            int kk0 = k0 + koff;
            bf0 = load_b_frag_n(Bb, ldB, kk0, n0 +  0 + colL);
            bf1 = load_b_frag_n(Bb, ldB, kk0, n0 + 16 + colL);
            bf2 = load_b_frag_n(Bb, ldB, kk0, n0 + 32 + colL);
            bf3 = load_b_frag_n(Bb, ldB, kk0, n0 + 48 + colL);
        }
        acc0 = __builtin_amdgcn_wmma_f32_16x16x32_f16(false, af, false, bf0, (short)0, acc0, false, false);
        acc1 = __builtin_amdgcn_wmma_f32_16x16x32_f16(false, af, false, bf1, (short)0, acc1, false, false);
        acc2 = __builtin_amdgcn_wmma_f32_16x16x32_f16(false, af, false, bf2, (short)0, acc2, false, false);
        acc3 = __builtin_amdgcn_wmma_f32_16x16x32_f16(false, af, false, bf3, (short)0, acc3, false, false);
    }

    v8f accs[4] = {acc0, acc1, acc2, acc3};
#pragma unroll
    for (int r = 0; r < 8; ++r) {
        int m = m0 + ((lane < 16) ? r : (r + 8));
        if (m < M) {
#pragma unroll
            for (int j = 0; j < 4; ++j) {
                int n = n0 + j * 16 + colL;
                float v = accs[j][r];
                if (bias) v += bias[n];
                if (act == 1) v = gelu_f(v);
                if (R) v += R[b * sR + (long)m * ldR + n];
                C[b * sC + (long)m * ldC + n] = v;
            }
        }
    }
}

// ---------------------------------------------------------------------------
// Scatter-add compact x1 rows back into the residual stream.
// ---------------------------------------------------------------------------
__global__ void k_scatter_add_x1(const float* __restrict__ src,
                                 float* __restrict__ y) {
    long i = (long)blockIdx.x * blockDim.x + threadIdx.x;
    if (i >= (long)22032 * 256) return;
    long tc = i >> 8;
    int d = (int)(i & 255);
    int b = (int)(tc / 1377);
    int t = (int)(tc % 1377);
    y[((long)b * 2754 + t) * 256 + d] += src[i];
}

// ---------------------------------------------------------------------------
// Prefill attention output buffer xo with depthwise-conv(V) + eps*emb terms.
// Channel packing after concat/transpose: out_ch = h*32 + c (+16 temporal).
// ---------------------------------------------------------------------------
__global__ void k_conv_prefill(const float* __restrict__ qkv,
                               const float* __restrict__ cw_s, const float* __restrict__ cb_s,
                               const float* __restrict__ cw_t, const float* __restrict__ cb_t,
                               const float* __restrict__ emb,
                               float* __restrict__ xo) {
    long i = (long)blockIdx.x * blockDim.x + threadIdx.x;
    if (i >= (long)44064 * 256) return;
    long token = i >> 8;
    int dfull = (int)(i & 255);
    int s = (int)(token % 17);
    long bt = token / 17;
    int t = (int)(bt % 162);
    int b = (int)(bt / 162);

    int isT = dfull >= 128;
    int dd = isT ? dfull - 128 : dfull;
    const float* w  = isT ? cw_t : cw_s;
    const float* cb = isT ? cb_t : cb_s;
    int col = 3 * (dd + (isT ? 128 : 0)) + 2;     // v channel in qkv row
    int tlo = (t < 81) ? 0 : 81;                  // conv split at ADF

    float acc = cb[dd];
    for (int kh = 0; kh < 3; ++kh) {
        int tt = t + kh - 1;
        if (tt < tlo || tt >= tlo + 81) continue;
        for (int kw = 0; kw < 3; ++kw) {
            int s2 = s + kw - 1;
            if (s2 < 0 || s2 >= 17) continue;
            acc += w[dd * 9 + kh * 3 + kw] *
                   qkv[((long)((b * 162 + tt) * 17 + s2)) * 768 + col];
        }
    }
    float eps = isT ? 1e-9f : 1e-4f;
    acc += eps * emb[d_PART[s] * 16 + (dd & 15)];

    int h = dd >> 4, c = dd & 15;
    xo[token * 256 + h * 32 + (isT ? 16 : 0) + c] = acc;
}

// ---------------------------------------------------------------------------
// Spatial attention: one wave per (b,h,t); 17 queries over 17 keys, dim 16.
// ---------------------------------------------------------------------------
__global__ void k_attn_spatial(const float* __restrict__ qkv,
                               float* __restrict__ xo) {
    __shared__ float Q[17][16], Kk[17][16], V[17][16];
    int g = blockIdx.x;
    int t = g % 162;
    int h = (g / 162) % 8;
    int b = g / (162 * 8);
    int lane = threadIdx.x;

    if (lane < 17) {
        long tok = (long)(b * 162 + t) * 17 + lane;
        for (int c = 0; c < 16; ++c) {
            long base = tok * 768 + 3 * (h * 16 + c);
            Q[lane][c]  = qkv[base];
            Kk[lane][c] = qkv[base + 1];
            V[lane][c]  = qkv[base + 2];
        }
    }
    __syncthreads();
    if (lane < 17) {
        const float scale = 0.08838834764831845f;   // 128^-0.5
        float sc[17], mx = -1e30f;
        for (int j = 0; j < 17; ++j) {
            float dot = 0.f;
            for (int c = 0; c < 16; ++c) dot += Q[lane][c] * Kk[j][c];
            sc[j] = dot * scale;
            mx = fmaxf(mx, sc[j]);
        }
        float sum = 0.f, out[16];
        for (int c = 0; c < 16; ++c) out[c] = 0.f;
        for (int j = 0; j < 17; ++j) {
            float e = expf(sc[j] - mx);
            sum += e;
            for (int c = 0; c < 16; ++c) out[c] += e * V[j][c];
        }
        float inv = 1.f / sum;
        long tok = (long)(b * 162 + t) * 17 + lane;
        for (int c = 0; c < 16; ++c)
            xo[tok * 256 + h * 32 + c] += out[c] * inv;
    }
}

// ---------------------------------------------------------------------------
// Temporal attention: one block per (b,h,s); 162 queries over 162 keys,
// dim 16, K/V staged in LDS.
// ---------------------------------------------------------------------------
__global__ void k_attn_temporal(const float* __restrict__ qkv,
                                float* __restrict__ xo) {
    __shared__ float Kt[162][16];
    __shared__ float Vt[162][16];
    int g = blockIdx.x;
    int s = g % 17;
    int h = (g / 17) % 8;
    int b = g / (17 * 8);
    int tid = threadIdx.x;

    if (tid < 162) {
        long tok = (long)(b * 162 + tid) * 17 + s;
        for (int c = 0; c < 16; ++c) {
            long base = tok * 768 + 3 * (128 + h * 16 + c);
            Kt[tid][c] = qkv[base + 1];
            Vt[tid][c] = qkv[base + 2];
        }
    }
    __syncthreads();
    if (tid < 162) {
        const float scale = 0.08838834764831845f;
        long tokq = (long)(b * 162 + tid) * 17 + s;
        float qv[16];
        for (int c = 0; c < 16; ++c)
            qv[c] = qkv[tokq * 768 + 3 * (128 + h * 16 + c)];
        float mx = -1e30f;
        for (int j = 0; j < 162; ++j) {
            float dot = 0.f;
            for (int c = 0; c < 16; ++c) dot += qv[c] * Kt[j][c];
            mx = fmaxf(mx, dot * scale);
        }
        float sum = 0.f, out[16];
        for (int c = 0; c < 16; ++c) out[c] = 0.f;
        for (int j = 0; j < 162; ++j) {
            float dot = 0.f;
            for (int c = 0; c < 16; ++c) dot += qv[c] * Kt[j][c];
            float e = expf(dot * scale - mx);
            sum += e;
            for (int c = 0; c < 16; ++c) out[c] += e * Vt[j][c];
        }
        float inv = 1.f / sum;
        for (int c = 0; c < 16; ++c)
            xo[tokq * 256 + h * 32 + 16 + c] += out[c] * inv;
    }
}

// ---------------------------------------------------------------------------
// Final head: out[b,q,s,j] = sum_c headtmp[b,q,s,c] * head_w[j,c]   (N=3)
// ---------------------------------------------------------------------------
__global__ void k_head_final(const float* __restrict__ ht,
                             const float* __restrict__ hw,
                             float* __restrict__ out) {
    int i = blockIdx.x * blockDim.x + threadIdx.x;
    if (i >= 16 * 243 * 17 * 3) return;
    int j = i % 3;
    int r = i / 3;
    int s = r % 17; r /= 17;
    int q = r % 243;
    int b = r / 243;
    const float* base = ht + ((long)(b * 243 + q) * 4352 + s * 256);
    float acc = 0.f;
    for (int c = 0; c < 256; ++c) acc += base[c] * hw[j * 256 + c];
    out[i] = acc;
}

// ---------------------------------------------------------------------------
// Host orchestration
// ---------------------------------------------------------------------------
extern "C" void kernel_launch(void* const* d_in, const int* in_sizes, int n_in,
                              void* d_out, int out_size, void* d_ws, size_t ws_size,
                              hipStream_t stream) {
    (void)in_sizes; (void)n_in; (void)out_size; (void)ws_size;

    const float* x          = (const float*)d_in[0];
    const float* pose_w     = (const float*)d_in[1];   // (256,2)
    const float* head_pre_w = (const float*)d_in[2];   // (243,162)
    const float* head_w     = (const float*)d_in[3];   // (3,256)
    auto BP = [&](int L, int idx) { return (const float*)d_in[4 + L * 19 + idx]; };
    // 0 ln1_g 1 ln1_b 2 qkv_w 3 qkv_b 4 proj_w 5 proj_b 6 emb 7 conv_s_w
    // 8 conv_s_b 9 conv_t_w 10 conv_t_b 11 ln2_g 12 ln2_b 13 fc1_w 14 fc2_w
    // 15 ffc1_w 16 ffc1_b 17 ffc2_w 18 ffc2_b

    // ---- workspace layout (floats) ----
    float* ws      = (float*)d_ws;
    float* w_dctf  = ws;                       // 81*243 (unpadded, VALU use only)
    float* w_dctk  = w_dctf + 19712;           // 96*96 padded
    float* w_idctk = w_dctk + 9216;            // 96*96 padded
    float* w_hpre  = w_idctk + 9216;           // 256*192 padded head_pre_w
    float* w_xc    = w_hpre + 49152;           // 44064*2
    float* w_y     = w_xc + 88192;             // 44064*256 residual stream
    float* w_xn    = w_y + 11280384;           // 44064*256: LN out / xo / compact
    float* w_qkv   = w_xn + 11280384;          // 44064*768: qkv / hidden / head tmp
    float* w_y2    = w_qkv + 33841152;         // 22032*256 compact scratch

    auto gemm = [&](const float* A, long ldA, long sA,
                    const float* Bp, long ldB, long sB, int transB,
                    const float* bias, const float* R, long ldR, long sR,
                    float* C, long ldC, long sC,
                    int M, int N, int Kpad, int act, int batch) {
        dim3 grid((M + 15) / 16, N / 64, batch);
        if (transB)
            k_gemm<1><<<grid, 32, 0, stream>>>(A, ldA, sA, Bp, ldB, sB,
                                               bias, R, ldR, sR, C, ldC, sC,
                                               M, N, Kpad, act);
        else
            k_gemm<0><<<grid, 32, 0, stream>>>(A, ldA, sA, Bp, ldB, sB,
                                               bias, R, ldR, sR, C, ldC, sC,
                                               M, N, Kpad, act);
    };

    // ---- constants / input embedding ----
    {
        int total = 81 * 243 + 96 * 96 + 96 * 96 + 256 * 192;  // 87267
        k_const_init<<<(total + 255) / 256, 256, 0, stream>>>(
            head_pre_w, w_dctf, w_dctk, w_idctk, w_hpre);
    }
    k_build_xc<<<(88128 + 255) / 256, 256, 0, stream>>>(x, w_dctf, w_xc);
    k_embed<<<44064, 256, 0, stream>>>(w_xc, pose_w, w_y);

    // ---- transformer blocks ----
    for (int L = 0; L < 3; ++L) {
        // ---- attention ----
        k_ln<<<44064, 256, 0, stream>>>(w_y, w_xn, BP(L,0), BP(L,1), 0);
        gemm(w_xn, 256, 0, BP(L,2), 256, 0, 1, BP(L,3),
             nullptr, 0, 0, w_qkv, 768, 0, 44064, 768, 256, 0, 1);
        k_conv_prefill<<<44064, 256, 0, stream>>>(w_qkv, BP(L,7), BP(L,8),
                                                  BP(L,9), BP(L,10), BP(L,6), w_xn);
        k_attn_spatial<<<16 * 8 * 162, 32, 0, stream>>>(w_qkv, w_xn);
        k_attn_temporal<<<16 * 8 * 17, 192, 0, stream>>>(w_qkv, w_xn);
        gemm(w_xn, 256, 0, BP(L,4), 256, 0, 1, BP(L,5),
             w_y, 256, 0, w_y, 256, 0, 44064, 256, 256, 0, 1);

        // ---- MLP on x1 (t<81), compact rows ----
        k_ln<<<22032, 256, 0, stream>>>(w_y, w_xn, BP(L,11), BP(L,12), 1);
        gemm(w_xn, 256, 0, BP(L,13), 256, 0, 1, nullptr,
             nullptr, 0, 0, w_qkv, 1024, 0, 22032, 1024, 256, 1, 1);
        gemm(w_qkv, 1024, 0, BP(L,14), 1024, 0, 1, nullptr,
             nullptr, 0, 0, w_y2, 256, 0, 22032, 256, 1024, 0, 1);
        k_scatter_add_x1<<<22032, 256, 0, stream>>>(w_y2, w_y);

        // ---- frequency branch on x2 (t>=81), compact rows ----
        k_ln<<<22032, 256, 0, stream>>>(w_y, w_xn, BP(L,11), BP(L,12), 2);
        // y2[b] = IDCT_K (81x81, padded 96x96) @ n2c[b] (81 x 4352)
        gemm(w_idctk, 96, 0, w_xn, 4352, 352512, 0, nullptr,
             nullptr, 0, 0, w_y2, 4352, 352512, 81, 4352, 96, 0, 16);
        gemm(w_y2, 256, 0, BP(L,15), 256, 0, 1, BP(L,16),
             nullptr, 0, 0, w_qkv, 1024, 0, 22032, 1024, 256, 1, 1);
        gemm(w_qkv, 1024, 0, BP(L,17), 1024, 0, 1, BP(L,18),
             nullptr, 0, 0, w_xn, 256, 0, 22032, 256, 1024, 0, 1);
        // x2 += DCT_K (padded 96x96) @ y3c[b] (81 x 4352)
        gemm(w_dctk, 96, 0, w_xn, 4352, 352512, 0, nullptr,
             w_y + 1377 * 256, 4352, 2754 * 256,
             w_y + 1377 * 256, 4352, 2754 * 256, 81, 4352, 96, 0, 16);
    }

    // ---- head ----
    // headtmp[b] (243 x 4352) = head_pre_w (padded 256x192) @ Y[b] (162 x 4352)
    gemm(w_hpre, 192, 0, w_y, 4352, 705024, 0, nullptr,
         nullptr, 0, 0, w_qkv, 4352, 1057536, 243, 4352, 192, 0, 16);
    k_head_final<<<(198288 + 255) / 256, 256, 0, stream>>>(w_qkv, head_w,
                                                           (float*)d_out);
}
